// MultiHeadSelfAttention_25400436588671
// MI455X (gfx1250) — compile-verified, matrix operand order checked by reading, not run on hardware
//
#include <hip/hip_runtime.h>
#include <hip/hip_bf16.h>
#include <math.h>

// ---------------- constants ----------------
#define BB 4
#define SS 2048
#define DD 1024
#define HH 16
#define HD 64
#define MM (BB * SS)          // 8192 rows of [M,D] activations
#define KCH 128               // K-chunk (bf16 elements) staged to LDS per buffer

typedef __attribute__((ext_vector_type(16))) __bf16 v16bf;
typedef __attribute__((ext_vector_type(8)))  __bf16 v8bf;
typedef __attribute__((ext_vector_type(8)))  float  v8f;
typedef __attribute__((ext_vector_type(4)))  float  v4f;

// ---------------- helpers ----------------
__device__ __forceinline__ __bf16 f2bf(float f) {
    unsigned int u = __builtin_bit_cast(unsigned int, f);
    u += 0x7FFFu + ((u >> 16) & 1u);           // RNE
    unsigned short h = (unsigned short)(u >> 16);
    return __builtin_bit_cast(__bf16, h);
}

// 16-element bf16 WMMA fragment from K-contiguous memory (ISA 7.12.2 layout):
// lanes 0-15 hold K=0..7 & 16..23, lanes 16-31 hold K=8..15 & 24..31.
__device__ __forceinline__ v16bf frag_ld(const __bf16* p, int half) {
    const v8bf* pv = (const v8bf*)p;
    v8bf c0 = pv[half];
    v8bf c1 = pv[2 + half];
    return __builtin_shufflevector(c0, c1, 0, 1, 2, 3, 4, 5, 6, 7,
                                   8, 9, 10, 11, 12, 13, 14, 15);
}

__device__ __forceinline__ v8f wmma_bf16(v16bf a, v16bf b, v8f c) {
    return __builtin_amdgcn_wmma_f32_16x16x32_bf16(false, a, false, b, (short)0, c,
                                                   false, false);
}

// CDNA5 async global->LDS copy, 16B per lane, tracked by ASYNCcnt.
// LDS offset = low 32 bits of the flat shared-memory address (ISA 10.2).
__device__ __forceinline__ void async_cp16(void* lds_ptr, const __bf16* gptr) {
    unsigned loff = (unsigned)(unsigned long long)lds_ptr;
    asm volatile("global_load_async_to_lds_b128 %0, %1, off"
                 :: "v"(loff), "v"(gptr) : "memory");
}
__device__ __forceinline__ void wait_async0() {
    asm volatile("s_wait_asynccnt 0x0" ::: "memory");
}

// Stage one 64(n) x KCH(k) bf16 weight tile into LDS (16KB = 4 async issues
// of 256 lanes x 16B). LDS layout: row-major [64][KCH].
__device__ __forceinline__ void stage_w(__bf16* sB, const __bf16* W, int n0,
                                        int k0, int tid) {
#pragma unroll
    for (int i = 0; i < 4; ++i) {
        const int idx = i * 256 + tid;          // 0..1023 16B chunks
        const int n = idx >> 4;                 // 0..63
        const int kb = (idx & 15) * 8;          // 0..120
        async_cp16(&sB[(size_t)idx * 8], W + (size_t)(n0 + n) * DD + k0 + kb);
    }
}

// ---------------- kernel 0: fp32 -> bf16 pre-conversion ----------------
__global__ __launch_bounds__(256) void cvt_bf16(const float* __restrict__ src,
                                                __bf16* __restrict__ dst, int n) {
    int i = (blockIdx.x * 256 + threadIdx.x) * 8;
    if (i >= n) return;
    v4f a = *(const v4f*)(src + i);
    v4f b = *(const v4f*)(src + i + 4);
    v8bf o;
#pragma unroll
    for (int e = 0; e < 4; ++e) { o[e] = f2bf(a[e]); o[e + 4] = f2bf(b[e]); }
    *(v8bf*)(dst + i) = o;
}

// ---------------- kernel 1: fused QKV projection ----------------
// grid: (MM/128, DD/64, 3), block 256 (8 waves). Block tile 128x64; wave tile
// 16x64 (4 accumulators). Weight tiles double-buffered in LDS via async copy.
// Q/K stored [B,H,S,HD]; V stored transposed [B,H,HD,S].
__global__ __launch_bounds__(256) void mhsa_qkv(const __bf16* __restrict__ xb,
                                                const __bf16* __restrict__ wq,
                                                const __bf16* __restrict__ wk,
                                                const __bf16* __restrict__ wv,
                                                const float* __restrict__ bq,
                                                const float* __restrict__ bk,
                                                const float* __restrict__ bv,
                                                __bf16* __restrict__ qkv_out) {
    __shared__ __align__(16) __bf16 sB[2 * 64 * KCH];   // 32 KB double buffer

    const int tid = threadIdx.x;
    const int w = tid >> 5, lane = tid & 31;
    const int half = lane >> 4, l16 = lane & 15;
    const int z = blockIdx.z;
    const __bf16* W    = (z == 0) ? wq : (z == 1) ? wk : wv;
    const float*  bias = (z == 0) ? bq : (z == 1) ? bk : bv;
    __bf16* dst = qkv_out + (size_t)z * MM * DD;

    const int m0 = blockIdx.x * 128 + w * 16;
    const int n0 = blockIdx.y * 64;
    const __bf16* arow = xb + (size_t)(m0 + l16) * DD;

    v8f acc[4] = {};
    stage_w(sB, W, n0, 0, tid);
    for (int c = 0; c < DD / KCH; ++c) {
        wait_async0();
        __syncthreads();
        if (c + 1 < DD / KCH) stage_w(sB + ((c + 1) & 1) * 64 * KCH, W, n0,
                                      (c + 1) * KCH, tid);
        const __bf16* sBuf = sB + (c & 1) * 64 * KCH;
#pragma unroll
        for (int kk = 0; kk < KCH; kk += 32) {
            v16bf a = frag_ld(arow + c * KCH + kk, half);
#pragma unroll
            for (int j = 0; j < 4; ++j) {
                v16bf b = frag_ld(&sBuf[(16 * j + l16) * KCH + kk], half);
                acc[j] = wmma_bf16(a, b, acc[j]);
            }
        }
        __syncthreads();   // done reading sBuf before it is refilled
    }

#pragma unroll
    for (int j = 0; j < 4; ++j) {
        const int ng = n0 + 16 * j + l16;
        const int h = ng >> 6, hd = ng & 63;
        const float bv_ = bias[ng];
#pragma unroll
        for (int r = 0; r < 8; ++r) {
            const int mg = m0 + r + 8 * half;
            const int bi = mg >> 11, s = mg & (SS - 1);
            const float v = acc[j][r] + bv_;
            if (z < 2)   // Q, K: [B,H,S,HD]
                dst[(((size_t)bi * HH + h) * SS + s) * HD + hd] = f2bf(v);
            else         // V: [B,H,HD,S] (transposed)
                dst[(((size_t)bi * HH + h) * HD + hd) * SS + s] = f2bf(v);
        }
    }
}

// ---------------- kernel 2: attention core ----------------
// grid: (SS/16, HH, BB), block 256 (8 waves). 16 query rows per block.
// LDS: 128KB fp32 scores + 64KB bf16 P = 192KB (CDNA5 320KB WGP LDS).
__global__ __launch_bounds__(256) void mhsa_attn(const __bf16* __restrict__ qkv,
                                                 float* __restrict__ attn_out,
                                                 __bf16* __restrict__ ctx_ws) {
    __shared__ float sS[16 * SS];                     // reused for split-K partials
    __shared__ __align__(16) __bf16 sP[16 * SS];

    const int w = threadIdx.x >> 5, lane = threadIdx.x & 31;
    const int half = lane >> 4, l16 = lane & 15;
    const int q0 = blockIdx.x * 16, h = blockIdx.y, b = blockIdx.z;
    const int bh = b * HH + h;

    const __bf16* Qhead  = qkv + (size_t)bh * SS * HD;
    const __bf16* Khead  = qkv + (size_t)1 * MM * DD + (size_t)bh * SS * HD;
    const __bf16* Vthead = qkv + (size_t)2 * MM * DD + (size_t)bh * HD * SS;

    // ---- phase 1: scores = (Q K^T) * scale ----
    const __bf16* qrow = Qhead + (size_t)(q0 + l16) * HD;
    const v16bf qf0 = frag_ld(qrow, half);
    const v16bf qf1 = frag_ld(qrow + 32, half);
    for (int jt = w; jt < SS / 16; jt += 8) {
        const int c0 = jt * 16;
        const __bf16* krow = Khead + (size_t)(c0 + l16) * HD;
        v8f acc = {};
        acc = wmma_bf16(qf0, frag_ld(krow, half), acc);
        acc = wmma_bf16(qf1, frag_ld(krow + 32, half), acc);
#pragma unroll
        for (int r = 0; r < 8; ++r)
            sS[(r + 8 * half) * SS + c0 + l16] = acc[r] * 0.125f;  // 1/sqrt(64)
    }
    __syncthreads();

    // ---- phase 2: row softmax (wave32 shuffles); write attn + bf16 P ----
#pragma unroll
    for (int rr = 0; rr < 2; ++rr) {
        const int row = w * 2 + rr;
        float* rp = &sS[row * SS];
        float mx = -INFINITY;
        for (int c = lane; c < SS; c += 32) mx = fmaxf(mx, rp[c]);
#pragma unroll
        for (int o = 16; o > 0; o >>= 1) mx = fmaxf(mx, __shfl_xor(mx, o, 32));
        float sum = 0.f;
        for (int c = lane; c < SS; c += 32) {
            float e = __expf(rp[c] - mx);
            rp[c] = e;
            sum += e;
        }
#pragma unroll
        for (int o = 16; o > 0; o >>= 1) sum += __shfl_xor(sum, o, 32);
        const float inv = 1.0f / sum;
        float* gout = attn_out + ((size_t)bh * SS + (q0 + row)) * SS;
        for (int c = lane; c < SS; c += 32) {
            const float p = rp[c] * inv;
            gout[c] = p;
            sP[row * SS + c] = f2bf(p);
        }
    }
    __syncthreads();

    // ---- phase 3: ctx = P @ V, split-K over 8 waves (4 n-tiles x 2 K-halves) ----
    {
        const int nt = w & 3, kh = w >> 2;
        const int n0 = nt * 16, kbase = kh * (SS / 2);
        const __bf16* vcol = Vthead + (size_t)(n0 + l16) * SS + kbase;  // K-contig
        const __bf16* prow = &sP[l16 * SS + kbase];
        v8f acc = {};
        for (int k0 = 0; k0 < SS / 2; k0 += 32) {
            v16bf a = frag_ld(prow + k0, half);     // ds_load_b128
            v16bf bfr = frag_ld(vcol + k0, half);   // global_load_b128
            acc = wmma_bf16(a, bfr, acc);
        }
        float* part = sS;
#pragma unroll
        for (int r = 0; r < 8; ++r) part[w * 256 + r * 32 + lane] = acc[r];
    }
    __syncthreads();
    if (w < 4) {
#pragma unroll
        for (int r = 0; r < 8; ++r) {
            const float v = sS[w * 256 + r * 32 + lane] + sS[(w + 4) * 256 + r * 32 + lane];
            const int m = r + 8 * half;
            ctx_ws[((size_t)b * SS + (q0 + m)) * DD + h * HD + w * 16 + l16] = f2bf(v);
        }
    }
}

// ---------------- kernel 3: output projection ----------------
// out = ctx @ Wo^T + bo (fp32). Same async-staged structure as mhsa_qkv.
__global__ __launch_bounds__(256) void mhsa_oproj(const __bf16* __restrict__ ctx_ws,
                                                  const __bf16* __restrict__ wo,
                                                  const float* __restrict__ bo,
                                                  float* __restrict__ out) {
    __shared__ __align__(16) __bf16 sB[2 * 64 * KCH];

    const int tid = threadIdx.x;
    const int w = tid >> 5, lane = tid & 31;
    const int half = lane >> 4, l16 = lane & 15;
    const int m0 = blockIdx.x * 128 + w * 16;
    const int n0 = blockIdx.y * 64;
    const __bf16* arow = ctx_ws + (size_t)(m0 + l16) * DD;

    v8f acc[4] = {};
    stage_w(sB, wo, n0, 0, tid);
    for (int c = 0; c < DD / KCH; ++c) {
        wait_async0();
        __syncthreads();
        if (c + 1 < DD / KCH) stage_w(sB + ((c + 1) & 1) * 64 * KCH, wo, n0,
                                      (c + 1) * KCH, tid);
        const __bf16* sBuf = sB + (c & 1) * 64 * KCH;
#pragma unroll
        for (int kk = 0; kk < KCH; kk += 32) {
            v16bf a = frag_ld(arow + c * KCH + kk, half);
#pragma unroll
            for (int j = 0; j < 4; ++j) {
                v16bf b = frag_ld(&sBuf[(16 * j + l16) * KCH + kk], half);
                acc[j] = wmma_bf16(a, b, acc[j]);
            }
        }
        __syncthreads();
    }

#pragma unroll
    for (int j = 0; j < 4; ++j) {
        const int ng = n0 + 16 * j + l16;
        const float bv_ = bo[ng];
#pragma unroll
        for (int r = 0; r < 8; ++r)
            out[(size_t)(m0 + r + 8 * half) * DD + ng] = acc[j][r] + bv_;
    }
}

// ---------------- launcher ----------------
extern "C" void kernel_launch(void* const* d_in, const int* in_sizes, int n_in,
                              void* d_out, int out_size, void* d_ws, size_t ws_size,
                              hipStream_t stream) {
    const float* x  = (const float*)d_in[0];
    const float* Wq = (const float*)d_in[1];
    const float* bq = (const float*)d_in[2];
    const float* Wk = (const float*)d_in[3];
    const float* bk = (const float*)d_in[4];
    const float* Wv = (const float*)d_in[5];
    const float* bv = (const float*)d_in[6];
    const float* Wo = (const float*)d_in[7];
    const float* bo = (const float*)d_in[8];

    float* out  = (float*)d_out;                       // [B,S,D]
    float* attn = (float*)d_out + (size_t)MM * DD;     // [B,H,S,S]

    __bf16* ws = (__bf16*)d_ws;
    const size_t NX = (size_t)MM * DD;                 // 8,388,608
    const size_t NW = (size_t)DD * DD;                 // 1,048,576
    __bf16* xb  = ws;
    __bf16* wqb = ws + NX;
    __bf16* wkb = wqb + NW;
    __bf16* wvb = wkb + NW;
    __bf16* wob = wvb + NW;
    __bf16* qkv = wob + NW;                            // Q, K, V(t): 3 * NX
    __bf16* ctx = qkv + 3 * NX;                        // NX

    cvt_bf16<<<(int)(NX / (8 * 256)), 256, 0, stream>>>(x, xb, (int)NX);
    cvt_bf16<<<(int)(NW / (8 * 256)), 256, 0, stream>>>(Wq, wqb, (int)NW);
    cvt_bf16<<<(int)(NW / (8 * 256)), 256, 0, stream>>>(Wk, wkb, (int)NW);
    cvt_bf16<<<(int)(NW / (8 * 256)), 256, 0, stream>>>(Wv, wvb, (int)NW);
    cvt_bf16<<<(int)(NW / (8 * 256)), 256, 0, stream>>>(Wo, wob, (int)NW);

    mhsa_qkv<<<dim3(MM / 128, DD / 64, 3), 256, 0, stream>>>(xb, wqb, wkb, wvb,
                                                             bq, bk, bv, qkv);
    mhsa_attn<<<dim3(SS / 16, HH, BB), 256, 0, stream>>>(qkv, attn, ctx);
    mhsa_oproj<<<dim3(MM / 128, DD / 64), 256, 0, stream>>>(ctx, wob, bo, out);
}
